// _P1QuantumClassifier_8160437862655
// MI455X (gfx1250) — matrix operand order repaired and checked by reading, not compiled, not run on hardware
//
#include <hip/hip_runtime.h>
#include <math.h>

typedef __attribute__((ext_vector_type(16))) _Float16 v16h;
typedef __attribute__((ext_vector_type(8)))  float    v8f;

#define NLAYERS 3

// ---------------- prep: build M = Re(U^H D U) from the 24 weights ----------------
// Wire i corresponds to index bit (8 >> i) of the flattened 16-dim state.

template<int BIT>
__device__ __forceinline__ void apply_ry(float c, float s, float* sr, float* si) {
#pragma unroll
  for (int j = 0; j < 16; ++j) {
    if (!(j & BIT)) {
      const int j1 = j | BIT;
      float r0 = sr[j], i0 = si[j], r1 = sr[j1], i1 = si[j1];
      sr[j]  = c * r0 - s * r1;   si[j]  = c * i0 - s * i1;
      sr[j1] = s * r0 + c * r1;   si[j1] = s * i0 + c * i1;
    }
  }
}

template<int BIT>
__device__ __forceinline__ void apply_rz(float cz, float sz, float* sr, float* si) {
#pragma unroll
  for (int j = 0; j < 16; ++j) {
    // bit=0: multiply by e^{-i t/2}; bit=1: e^{+i t/2}
    float ph = (j & BIT) ? sz : -sz;
    float r = sr[j], m = si[j];
    sr[j] = cz * r - ph * m;
    si[j] = cz * m + ph * r;
  }
}

template<int CBIT, int TBIT>
__device__ __forceinline__ void apply_cnot(float* sr, float* si) {
#pragma unroll
  for (int j = 0; j < 16; ++j) {
    if ((j & CBIT) && !(j & TBIT)) {
      const int j1 = j | TBIT;
      float tr = sr[j]; sr[j] = sr[j1]; sr[j1] = tr;
      float ti = si[j]; si[j] = si[j1]; si[j1] = ti;
    }
  }
}

__global__ void build_M_kernel(const float* __restrict__ w, float* __restrict__ M) {
  __shared__ float ure[16][16];   // [row m][column j] of U
  __shared__ float uim[16][16];
  const int t = threadIdx.x;

  if (t < 16) {
    // simulate basis state e_t through the 3 weight layers
    float sr[16], si[16];
#pragma unroll
    for (int j = 0; j < 16; ++j) { sr[j] = (j == t) ? 1.f : 0.f; si[j] = 0.f; }

#pragma unroll
    for (int layer = 0; layer < NLAYERS; ++layer) {
      const int base = layer * 8;
      {
        float h = 0.5f * w[base + 0], hz = 0.5f * w[base + 4];
        apply_ry<8>(__cosf(h), __sinf(h), sr, si);
        apply_rz<8>(__cosf(hz), __sinf(hz), sr, si);
      }
      {
        float h = 0.5f * w[base + 1], hz = 0.5f * w[base + 5];
        apply_ry<4>(__cosf(h), __sinf(h), sr, si);
        apply_rz<4>(__cosf(hz), __sinf(hz), sr, si);
      }
      {
        float h = 0.5f * w[base + 2], hz = 0.5f * w[base + 6];
        apply_ry<2>(__cosf(h), __sinf(h), sr, si);
        apply_rz<2>(__cosf(hz), __sinf(hz), sr, si);
      }
      {
        float h = 0.5f * w[base + 3], hz = 0.5f * w[base + 7];
        apply_ry<1>(__cosf(h), __sinf(h), sr, si);
        apply_rz<1>(__cosf(hz), __sinf(hz), sr, si);
      }
      apply_cnot<8, 4>(sr, si);   // CNOT(0,1)
      apply_cnot<4, 2>(sr, si);   // CNOT(1,2)
      apply_cnot<2, 1>(sr, si);   // CNOT(2,3)
      apply_cnot<1, 8>(sr, si);   // CNOT(3,0)
    }
#pragma unroll
    for (int m = 0; m < 16; ++m) { ure[m][t] = sr[m]; uim[m][t] = si[m]; }
  }
  __syncthreads();

  // M[a][b] = sum_m sign(m) * Re( conj(U[m][a]) * U[m][b] ), sign by bit3 of m
  const int a = t >> 4, b = t & 15;
  float acc = 0.f;
#pragma unroll
  for (int m = 0; m < 16; ++m) {
    float sgn = (m < 8) ? 1.f : -1.f;
    acc += sgn * (ure[m][a] * ure[m][b] + uim[m][a] * uim[m][b]);
  }
  M[a * 16 + b] = acc;
}

// ---------------- main: out[b] = s_b^T M s_b via v_wmma_f32_16x16x32_f16 ----------------

__global__ void qml_wmma_kernel(const float* __restrict__ x,
                                const float* __restrict__ M,
                                float* __restrict__ out,
                                int ntiles) {
  const int lane = threadIdx.x & 31;
  const int col  = lane & 15;
  const bool lo  = lane < 16;
  const int wave   = (blockIdx.x * blockDim.x + threadIdx.x) >> 5;
  const int nwaves = (gridDim.x * blockDim.x) >> 5;

  // A operand = M (symmetric) in 16-bit A 16x32 layout, K=16..31 hard zero.
  // lane L<16 : row L of M^T, elements 0..7 = M[0..7][L]
  // lane L+16 : row L of M^T, elements 0..7 = M[8..15][L]
  v16h amat = {};
  {
    const int rbase = lo ? 0 : 8;
#pragma unroll
    for (int k = 0; k < 8; ++k)
      amat[k] = (_Float16)M[(rbase + k) * 16 + col];
  }

  const float4* __restrict__ x4 = (const float4*)x;
  constexpr float HPI = 1.5707963267948966f;

  for (int tile = wave; tile < ntiles; tile += nwaves) {
    const int sample = tile * 16 + col;
    const float4 xv = x4[sample];

    // encoding state s = kron of (cos(pi x_i/2), sin(pi x_i/2)), wire 0 = bit 3.
    // x in [0,1) -> angle in [0, pi/2): hardware-native trig path is fine.
    const float a0[2] = {__cosf(HPI * xv.x), __sinf(HPI * xv.x)};
    const float a1[2] = {__cosf(HPI * xv.y), __sinf(HPI * xv.y)};
    const float a2[2] = {__cosf(HPI * xv.z), __sinf(HPI * xv.z)};
    const float a3[2] = {__cosf(HPI * xv.w), __sinf(HPI * xv.w)};

    // 24-multiply Kronecker tree: sv[q0q1q2q3] = (a0*a1)[hi] * (a2*a3)[lo]
    float t01[4], t23[4], sv[16];
#pragma unroll
    for (int j = 0; j < 4; ++j) {
      t01[j] = a0[(j >> 1) & 1] * a1[j & 1];
      t23[j] = a2[(j >> 1) & 1] * a3[j & 1];
    }
#pragma unroll
    for (int j = 0; j < 16; ++j)
      sv[j] = t01[j >> 2] * t23[j & 3];

    // B operand = S^T: lane b<16 holds s_b[0..7] in elems 0..7, lane b+16 holds s_b[8..15] in elems 0..7.
    // Elems 8..15 only multiply amat[8..15]==0: no mask needed.
    v16h bmat;
#pragma unroll
    for (int k = 0; k < 16; ++k)
      bmat[k] = (_Float16)(lo ? sv[k] : sv[k ^ 8]);

    v8f acc = {};
    acc = __builtin_amdgcn_wmma_f32_16x16x32_f16(
        /*neg_a=*/false, amat, /*neg_b=*/false, bmat,
        /*c_mod=*/(short)0, acc, /*reuse_a=*/false, /*reuse_b=*/false);

    // lane b   : acc[r] = (M s_b)[r]     -> needs dot with s_b[0..7]
    // lane b+16: acc[r] = (M s_b)[r+8]   -> needs dot with s_b[8..15]
    // Compute BOTH half-dots with constant indices (no dynamic extract!),
    // select once at the end.
    float pl = 0.f, ph = 0.f;
#pragma unroll
    for (int r = 0; r < 8; ++r) {
      pl = fmaf(sv[r],     acc[r], pl);
      ph = fmaf(sv[8 + r], acc[r], ph);
    }
    float partial = lo ? pl : ph;
    partial += __shfl_xor(partial, 16);
    if (lo) out[tile * 16 + col] = partial;
  }
}

extern "C" void kernel_launch(void* const* d_in, const int* in_sizes, int n_in,
                              void* d_out, int out_size, void* d_ws, size_t ws_size,
                              hipStream_t stream) {
  const float* x = (const float*)d_in[0];     // (BATCH, 4) float32
  const float* w = (const float*)d_in[1];     // (24,) float32
  float* out = (float*)d_out;                 // (BATCH,) float32
  float* M   = (float*)d_ws;                  // 256 floats scratch

  const int batch  = in_sizes[0] / 4;
  const int ntiles = batch / 16;              // 524288/16 = 32768

  build_M_kernel<<<1, 256, 0, stream>>>(w, M);
  qml_wmma_kernel<<<1024, 256, 0, stream>>>(x, M, out, ntiles);
}
